// Set2SetNet_36816459661882
// MI455X (gfx1250) — matrix operand order repaired and checked by reading, not compiled, not run on hardware
//
#include <hip/hip_runtime.h>
#include <hip/hip_bf16.h>
#include <math.h>

typedef __attribute__((ext_vector_type(16))) _Float16 v16h;
typedef __attribute__((ext_vector_type(8)))  _Float16 v8h;
typedef __attribute__((ext_vector_type(8)))  float    v8f;

#define N_NODES 50000
#define N_EDGES 800000
#define F 128
#define NGRAPH 512
#define KT 4        // K/32 for the 128-deep GCN GEMMs

static __device__ __forceinline__ float sigf(float x) { return 1.0f / (1.0f + expf(-x)); }

__global__ void k_fill_f32(float* p, float v, int n) {
    int i = blockIdx.x * blockDim.x + threadIdx.x;
    if (i < n) p[i] = v;
}

__global__ void k_cast_f16(const float* __restrict__ s, _Float16* __restrict__ d, int n) {
    int i = blockIdx.x * blockDim.x + threadIdx.x;
    if (i < n) d[i] = (_Float16)s[i];
}

__global__ void k_deg(const int* __restrict__ col, float* __restrict__ deg, int e) {
    int i = blockIdx.x * blockDim.x + threadIdx.x;
    if (i < e) atomicAdd(&deg[col[i]], 1.0f);
}

__global__ void k_dis(float* __restrict__ d, int n) {
    int i = blockIdx.x * blockDim.x + threadIdx.x;
    if (i < n) d[i] = rsqrtf(d[i] + 1.0f);   // +1 for self loop; deg>=1 always
}

// Pack W[K=128, N=128] (f32 row-major) into WMMA B-fragment order (f16):
// Bpack[(((ntile*KT + kti)*32 + lane)*16 + j] = (f16) W[(kti*32 + (lane>>4)*16 + j)*128 + ntile*16 + (lane&15)]
__global__ void k_pack_b(const float* __restrict__ Wf, _Float16* __restrict__ Bp) {
    int idx = blockIdx.x * blockDim.x + threadIdx.x;   // 128*128 = 16384
    if (idx >= F * F) return;
    int j     = idx & 15;
    int lane  = (idx >> 4) & 31;
    int kti   = (idx >> 9) & (KT - 1);
    int ntile = idx >> 11;
    int k = kti * 32 + (lane >> 4) * 16 + j;
    int n = ntile * 16 + (lane & 15);
    Bp[idx] = (_Float16)Wf[(size_t)k * F + n];
}

// C[M,128] = A[M,128] (f16 row-major) x B (pre-packed fragments), f32 accumulate.
// One wave per 16x16 tile. M%16==0.
__global__ void k_gemm_wmma(const _Float16* __restrict__ A,
                            const _Float16* __restrict__ Bp,
                            float* __restrict__ C) {
    const int lane = threadIdx.x;          // 0..31
    const int l    = lane & 15;
    const int hi   = lane >> 4;            // lane half select
    const int m0   = blockIdx.x * 16;
    const int n0   = blockIdx.y * 16;

    v8f acc = {};
    const _Float16* arow = A + (size_t)(m0 + l) * F;
    const _Float16* bbase = Bp + ((size_t)blockIdx.y * KT * 32 + lane) * 16;
#pragma unroll
    for (int kti = 0; kti < KT; ++kti) {
        // A fragment (contiguous K order, matching packed B):
        //  lanes 0-15 : elems 0-15 = K kt+0..15
        //  lanes 16-31: elems 0-15 = K kt+16..31
        const _Float16* ap = arow + kti * 32 + hi * 16;
        v8h lo  = *(const v8h*)(ap);
        v8h hi8 = *(const v8h*)(ap + 8);
        v16h av = __builtin_shufflevector(lo, hi8, 0, 1, 2, 3, 4, 5, 6, 7,
                                          8, 9, 10, 11, 12, 13, 14, 15);
        v16h bv = *(const v16h*)(bbase + (size_t)kti * 32 * 16);
        acc = __builtin_amdgcn_wmma_f32_16x16x32_f16(false, av, false, bv,
                                                     (short)0, acc, false, false);
    }
    // C layout: VGPR r -> row m0 + r + 8*hi, col n0 + l
#pragma unroll
    for (int r = 0; r < 8; ++r)
        C[(size_t)(m0 + r + hi * 8) * F + n0 + l] = acc[r];
}

// acc[col[e]] += dis[row]*dis[col] * xw[row]; self loops at e >= E.
// One thread per (edge, 4-feature group).
__global__ void k_scatter(const int* __restrict__ row, const int* __restrict__ col,
                          const float* __restrict__ dis, const float* __restrict__ xw,
                          float* __restrict__ acc, int E, int n) {
    int idx = blockIdx.x * blockDim.x + threadIdx.x;
    int total = (E + n) * (F / 4);
    if (idx >= total) return;
    int fq = idx & (F / 4 - 1);     // feature quad 0..31
    int e  = idx >> 5;
    int r, c;
    if (e < E) { r = row[e]; c = col[e]; }
    else       { r = c = e - E; }
    float nrm = dis[r] * dis[c];
    float4 v = ((const float4*)(xw + (size_t)r * F))[fq];
    float* dst = acc + (size_t)c * F + fq * 4;
    atomicAdd(dst + 0, nrm * v.x);
    atomicAdd(dst + 1, nrm * v.y);
    atomicAdd(dst + 2, nrm * v.z);
    atomicAdd(dst + 3, nrm * v.w);
}

__global__ void k_bias_relu(const float* __restrict__ acc, const float* __restrict__ b,
                            float* __restrict__ h32, _Float16* __restrict__ h16, int n) {
    int i = blockIdx.x * blockDim.x + threadIdx.x;
    if (i >= n) return;
    float v = acc[i] + b[i & (F - 1)];
    v = fmaxf(v, 0.0f);
    h32[i] = v;
    h16[i] = (_Float16)v;
}

// LSTM gates: gates[g,j] = bih[j]+bhh[j] + q_star[g,:]·Wih[j,:] + hs[g,:]·Whh[j,:]
__global__ void k_gates(const float* __restrict__ qstar, const float* __restrict__ hs,
                        const float* __restrict__ Wih, const float* __restrict__ Whh,
                        const float* __restrict__ bih, const float* __restrict__ bhh,
                        float* __restrict__ gates) {
    int idx = blockIdx.x * blockDim.x + threadIdx.x;
    if (idx >= NGRAPH * 4 * F) return;
    int g = idx >> 9;          // /512
    int j = idx & 511;
    float s = bih[j] + bhh[j];
    const float* q = qstar + (size_t)g * 2 * F;
    const float* wi = Wih + (size_t)j * 2 * F;
#pragma unroll 4
    for (int k = 0; k < 2 * F; ++k) s += q[k] * wi[k];
    const float* h = hs + (size_t)g * F;
    const float* wh = Whh + (size_t)j * F;
#pragma unroll 4
    for (int k = 0; k < F; ++k) s += h[k] * wh[k];
    gates[idx] = s;
}

__global__ void k_lstm(const float* __restrict__ gates, float* __restrict__ cs,
                       float* __restrict__ hs) {
    int idx = blockIdx.x * blockDim.x + threadIdx.x;
    if (idx >= NGRAPH * F) return;
    int g = idx >> 7, j = idx & (F - 1);
    const float* gr = gates + (size_t)g * 4 * F;
    float ig = sigf(gr[j]);
    float fg = sigf(gr[F + j]);
    float gg = tanhf(gr[2 * F + j]);
    float og = sigf(gr[3 * F + j]);
    float c = fg * cs[idx] + ig * gg;
    cs[idx] = c;
    hs[idx] = og * tanhf(c);
}

__global__ void k_attn_e(const float* __restrict__ h, const float* __restrict__ hs,
                         const int* __restrict__ batch, float* __restrict__ e, int n) {
    int i = blockIdx.x * blockDim.x + threadIdx.x;
    if (i >= n) return;
    const float* hp = h + (size_t)i * F;
    const float* qp = hs + (size_t)batch[i] * F;
    float s = 0.0f;
#pragma unroll 4
    for (int k = 0; k < F; ++k) s += hp[k] * qp[k];
    e[i] = s;
}

static __device__ __forceinline__ void atomicMaxFloat(float* addr, float val) {
    if (val >= 0.0f) atomicMax((int*)addr, __float_as_int(val));
    else             atomicMin((unsigned int*)addr, __float_as_uint(val));
}

__global__ void k_segmax(const float* __restrict__ e, const int* __restrict__ batch,
                         float* __restrict__ amax, int n) {
    int i = blockIdx.x * blockDim.x + threadIdx.x;
    if (i < n) atomicMaxFloat(&amax[batch[i]], e[i]);
}

__global__ void k_fixmax(float* __restrict__ amax, int n) {
    int i = blockIdx.x * blockDim.x + threadIdx.x;
    if (i < n && !isfinite(amax[i])) amax[i] = 0.0f;
}

__global__ void k_expsum(const float* __restrict__ e, const int* __restrict__ batch,
                         const float* __restrict__ amax, float* __restrict__ a,
                         float* __restrict__ asum, int n) {
    int i = blockIdx.x * blockDim.x + threadIdx.x;
    if (i >= n) return;
    int g = batch[i];
    float v = expf(e[i] - amax[g]);
    a[i] = v;
    atomicAdd(&asum[g], v);
}

// r[g] += (a[i]/den) * h[i]; one thread per (node, 4-feature group)
__global__ void k_r(const float* __restrict__ a, const float* __restrict__ h,
                    const int* __restrict__ batch, const float* __restrict__ asum,
                    float* __restrict__ r, int n) {
    int idx = blockIdx.x * blockDim.x + threadIdx.x;
    if (idx >= n * (F / 4)) return;
    int i = idx >> 5, fq = idx & (F / 4 - 1);
    int g = batch[i];
    float den = asum[g];
    den = den > 0.0f ? den : 1.0f;
    float w = a[i] / den;
    float4 v = ((const float4*)(h + (size_t)i * F))[fq];
    float* dst = r + (size_t)g * F + fq * 4;
    atomicAdd(dst + 0, w * v.x);
    atomicAdd(dst + 1, w * v.y);
    atomicAdd(dst + 2, w * v.z);
    atomicAdd(dst + 3, w * v.w);
}

__global__ void k_qstar(const float* __restrict__ hs, const float* __restrict__ r,
                        float* __restrict__ qstar) {
    int idx = blockIdx.x * blockDim.x + threadIdx.x;
    if (idx >= NGRAPH * F) return;
    int g = idx >> 7, j = idx & (F - 1);
    qstar[(size_t)g * 2 * F + j] = hs[idx];
    qstar[(size_t)g * 2 * F + F + j] = r[idx];
}

__global__ void k_mlp1(const float* __restrict__ qstar, const float* __restrict__ Wm,
                       const float* __restrict__ b, float* __restrict__ y) {
    int idx = blockIdx.x * blockDim.x + threadIdx.x;     // 512*128
    if (idx >= NGRAPH * F) return;
    int g = idx >> 7, j = idx & (F - 1);
    float s = b[j];
    const float* q = qstar + (size_t)g * 2 * F;
#pragma unroll 4
    for (int k = 0; k < 2 * F; ++k) s += q[k] * Wm[(size_t)k * F + j];
    y[idx] = fmaxf(s, 0.0f);
}

__global__ void k_mlp2(const float* __restrict__ y1, const float* __restrict__ Wm,
                       const float* __restrict__ b, float* __restrict__ y) {
    int idx = blockIdx.x * blockDim.x + threadIdx.x;     // 512*64
    if (idx >= NGRAPH * 64) return;
    int g = idx >> 6, j = idx & 63;
    float s = b[j];
    const float* yp = y1 + (size_t)g * F;
#pragma unroll 4
    for (int k = 0; k < F; ++k) s += yp[k] * Wm[(size_t)k * 64 + j];
    y[idx] = fmaxf(s, 0.0f);
}

__global__ void k_mlp3_lsm(const float* __restrict__ y2, const float* __restrict__ Wm,
                           const float* __restrict__ b, float* __restrict__ out) {
    int g = blockIdx.x * blockDim.x + threadIdx.x;
    if (g >= NGRAPH) return;
    float z[10];
#pragma unroll
    for (int c = 0; c < 10; ++c) {
        float s = b[c];
        for (int k = 0; k < 64; ++k) s += y2[(size_t)g * 64 + k] * Wm[(size_t)k * 10 + c];
        z[c] = s;
    }
    float m = z[0];
#pragma unroll
    for (int c = 1; c < 10; ++c) m = fmaxf(m, z[c]);
    float se = 0.0f;
#pragma unroll
    for (int c = 0; c < 10; ++c) se += expf(z[c] - m);
    float lse = m + logf(se);
#pragma unroll
    for (int c = 0; c < 10; ++c) out[(size_t)g * 10 + c] = z[c] - lse;
}

extern "C" void kernel_launch(void* const* d_in, const int* in_sizes, int n_in,
                              void* d_out, int out_size, void* d_ws, size_t ws_size,
                              hipStream_t stream) {
    const float* x     = (const float*)d_in[0];
    const int*   ei    = (const int*)d_in[1];     // [2, E]: row then col
    const int*   batch = (const int*)d_in[2];
    const float* Wl[3] = {(const float*)d_in[3], (const float*)d_in[5], (const float*)d_in[7]};
    const float* bl[3] = {(const float*)d_in[4], (const float*)d_in[6], (const float*)d_in[8]};
    const float* Wih = (const float*)d_in[9];
    const float* Whh = (const float*)d_in[10];
    const float* bih = (const float*)d_in[11];
    const float* bhh = (const float*)d_in[12];
    const float* L1w = (const float*)d_in[13];
    const float* L1b = (const float*)d_in[14];
    const float* L2w = (const float*)d_in[15];
    const float* L2b = (const float*)d_in[16];
    const float* L3w = (const float*)d_in[17];
    const float* L3b = (const float*)d_in[18];

    const int N = N_NODES, E = N_EDGES;
    const int* row = ei;
    const int* col = ei + E;

    // ---- workspace carve-out (256B aligned) ----
    char* w = (char*)d_ws;
    auto carve = [&](size_t bytes) {
        void* p = (void*)w;
        w += (bytes + 255) & ~(size_t)255;
        return p;
    };
    _Float16* h16  = (_Float16*)carve((size_t)N * F * 2);
    _Float16* bpack = (_Float16*)carve((size_t)F * F * 2);
    float* xw      = (float*)carve((size_t)N * F * 4);   // GEMM out; later aliased as final h32
    float* acc     = (float*)carve((size_t)N * F * 4);   // scatter accumulator
    float* dis     = (float*)carve((size_t)N * 4);
    float* ebuf    = (float*)carve((size_t)N * 4);
    float* abuf    = (float*)carve((size_t)N * 4);
    float* amax    = (float*)carve((size_t)NGRAPH * 4);
    float* asum    = (float*)carve((size_t)NGRAPH * 4);
    float* hs      = (float*)carve((size_t)NGRAPH * F * 4);
    float* cs      = (float*)carve((size_t)NGRAPH * F * 4);
    float* rbuf    = (float*)carve((size_t)NGRAPH * F * 4);
    float* qstar   = (float*)carve((size_t)NGRAPH * 2 * F * 4);
    float* gates   = (float*)carve((size_t)NGRAPH * 4 * F * 4);
    float* y1      = (float*)carve((size_t)NGRAPH * F * 4);
    float* y2      = (float*)carve((size_t)NGRAPH * 64 * 4);
    (void)ws_size; (void)n_in; (void)in_sizes; (void)out_size;

    auto blocks = [](int n) { return dim3((unsigned)((n + 255) / 256)); };

    // ---- degrees / symmetric norm ----
    k_fill_f32<<<blocks(N), 256, 0, stream>>>(dis, 0.0f, N);
    k_deg<<<blocks(E), 256, 0, stream>>>(col, dis, E);
    k_dis<<<blocks(N), 256, 0, stream>>>(dis, N);

    // ---- input features to f16 ----
    k_cast_f16<<<blocks(N * F), 256, 0, stream>>>(x, h16, N * F);

    // ---- 3 GCN layers ----
    for (int l = 0; l < 3; ++l) {
        k_pack_b<<<blocks(F * F), 256, 0, stream>>>(Wl[l], bpack);
        dim3 gg(N / 16, F / 16);
        k_gemm_wmma<<<gg, 32, 0, stream>>>(h16, bpack, xw);
        k_fill_f32<<<blocks(N * F), 256, 0, stream>>>(acc, 0.0f, N * F);
        int tot = (E + N) * (F / 4);
        k_scatter<<<blocks(tot), 256, 0, stream>>>(row, col, dis, xw, acc, E, N);
        // xw is dead after the scatter of this layer -> reuse as h32
        k_bias_relu<<<blocks(N * F), 256, 0, stream>>>(acc, bl[l], xw, h16, N * F);
    }
    float* h32 = xw;   // final node embeddings [N,128] fp32

    // ---- Set2Set (4 processing steps) ----
    k_fill_f32<<<blocks(NGRAPH * F), 256, 0, stream>>>(hs, 0.0f, NGRAPH * F);
    k_fill_f32<<<blocks(NGRAPH * F), 256, 0, stream>>>(cs, 0.0f, NGRAPH * F);
    k_fill_f32<<<blocks(NGRAPH * 2 * F), 256, 0, stream>>>(qstar, 0.0f, NGRAPH * 2 * F);
    for (int s = 0; s < 4; ++s) {
        k_gates<<<blocks(NGRAPH * 4 * F), 256, 0, stream>>>(qstar, hs, Wih, Whh, bih, bhh, gates);
        k_lstm<<<blocks(NGRAPH * F), 256, 0, stream>>>(gates, cs, hs);
        k_attn_e<<<blocks(N), 256, 0, stream>>>(h32, hs, batch, ebuf, N);
        k_fill_f32<<<blocks(NGRAPH), 256, 0, stream>>>(amax, -INFINITY, NGRAPH);
        k_fill_f32<<<blocks(NGRAPH), 256, 0, stream>>>(asum, 0.0f, NGRAPH);
        k_fill_f32<<<blocks(NGRAPH * F), 256, 0, stream>>>(rbuf, 0.0f, NGRAPH * F);
        k_segmax<<<blocks(N), 256, 0, stream>>>(ebuf, batch, amax, N);
        k_fixmax<<<blocks(NGRAPH), 256, 0, stream>>>(amax, NGRAPH);
        k_expsum<<<blocks(N), 256, 0, stream>>>(ebuf, batch, amax, abuf, asum, N);
        k_r<<<blocks(N * (F / 4)), 256, 0, stream>>>(abuf, h32, batch, asum, rbuf, N);
        k_qstar<<<blocks(NGRAPH * F), 256, 0, stream>>>(hs, rbuf, qstar);
    }

    // ---- MLP head + log_softmax ----
    k_mlp1<<<blocks(NGRAPH * F), 256, 0, stream>>>(qstar, L1w, L1b, y1);
    k_mlp2<<<blocks(NGRAPH * 64), 256, 0, stream>>>(y1, L2w, L2b, y2);
    k_mlp3_lsm<<<blocks(NGRAPH), 256, 0, stream>>>(y2, L3w, L3b, (float*)d_out);
}
